// SelfAttention_49709951484770
// MI455X (gfx1250) — matrix operand order repaired and checked by reading, not compiled, not run on hardware
//
#include <hip/hip_runtime.h>

#define B_  4
#define C_  256
#define S_  4096
#define NH_ 4
#define HD_ 64
#define C3_ 768
#define LDB 264   // LDS B-panel row stride in halves (528B = 512B row + 16B pad)

typedef __attribute__((ext_vector_type(16))) _Float16 v16h;
typedef __attribute__((ext_vector_type(8)))  _Float16 v8h;
typedef __attribute__((ext_vector_type(8)))  float    v8f;
typedef __attribute__((ext_vector_type(4)))  int      v4i_;
typedef __attribute__((ext_vector_type(4)))  unsigned u4i_;
typedef __attribute__((ext_vector_type(8)))  int      v8i_;

union Frag16 { v16h v; v8h h[2]; };

#if defined(__has_builtin)
#if __has_builtin(__builtin_amdgcn_global_load_async_to_lds_b128)
#define HAVE_ASYNC_LDS 1
#endif
#if __has_builtin(__builtin_amdgcn_tensor_load_to_lds)
#define HAVE_TDM 1
#endif
#endif

__device__ __forceinline__ float bcast_lane(float x, int srclane) {
  return __int_as_float(__builtin_amdgcn_ds_bpermute(srclane << 2, __float_as_int(x)));
}

// ---- async global->LDS copy path (ASYNCcnt) -------------------------------
__device__ __forceinline__ void copy16_g2l(const _Float16* g, _Float16* l) {
#if defined(HAVE_ASYNC_LDS)
  __builtin_amdgcn_global_load_async_to_lds_b128(
      (__attribute__((address_space(1))) v4i_*)g,
      (__attribute__((address_space(3))) v4i_*)l, 0, 0);
#else
  *(v8h*)l = *(const v8h*)g;
#endif
}

__device__ __forceinline__ void async_drain() {
#if defined(HAVE_ASYNC_LDS)
  asm volatile("s_wait_asynccnt 0x0" ::: "memory");
#endif
}

// stage 64 rows x 512B panel -> LDS rows of LDB halves, all 256 threads
__device__ __forceinline__ void stage_panel_async(const _Float16* BT, _Float16* ldsB, int t) {
#pragma unroll
  for (int i = 0; i < 8; ++i) {
    int chunk = t + 256 * i;
    int nn = chunk >> 5;            // 32 x 16B chunks per 512B row
    int ko = (chunk & 31) * 8;      // halves
    copy16_g2l(BT + (size_t)nn * 256 + ko, &ldsB[nn * LDB + ko]);
  }
  async_drain();
}

// ---- TDM path (TENSORcnt): one wave issues a 2-D padded descriptor --------
__device__ __forceinline__ void stage_panel_tdm(const _Float16* BT, _Float16* ldsB, int t) {
#if defined(HAVE_TDM)
  if ((t >> 5) == 0) {              // wave 0 only; TDM ignores EXEC
    unsigned long long ga = (unsigned long long)BT;
    unsigned lds = (unsigned)(unsigned long long)ldsB;
    // D# group 0: count=1, lds_addr, global_addr[56:0], type=2
    u4i_ g0 = { 1u,
                lds,
                (unsigned)ga,
                (unsigned)((ga >> 32) & 0x1FFFFFFu) | (2u << 30) };
    // D# group 1: data_size=2B, pad_enable, pad every 128 dwords by 4 dwords,
    // tensor = tile = 256(k) x 64(n), dim0_stride = 256 elements
    v8i_ g1 = { (int)((1u << 16) | (1u << 20) | (6u << 22) | (3u << 25)),
                (int)(256u << 16),   // tensor_dim0 lo16 at [31:16]
                (int)(64u << 16),    // dim0 hi=0 | tensor_dim1 lo16
                (int)(256u << 16),   // dim1 hi=0 | tile_dim0=256
                64,                  // tile_dim1=64, tile_dim2=0
                256,                 // tensor_dim0_stride lo32
                0, 0 };
    v4i_ gz = { 0, 0, 0, 0 };
    v8i_ gz8 = { 0, 0, 0, 0, 0, 0, 0, 0 };
    __builtin_amdgcn_tensor_load_to_lds(g0, g1, gz, gz, gz8, 0);
#if __has_builtin(__builtin_amdgcn_s_wait_tensorcnt)
    __builtin_amdgcn_s_wait_tensorcnt((short)0);
#else
    asm volatile("s_wait_tensorcnt 0x0" ::: "memory");
#endif
  }
#else
  stage_panel_async(BT, ldsB, t);
#endif
}

// ---------------------------------------------------------------------------
// Prep kernels: one-time f32->f16 convert + transpose (memory-bound).
// ---------------------------------------------------------------------------
__global__ __launch_bounds__(256) void transpose_x_kernel(
    const float* __restrict__ x, _Float16* __restrict__ xT)
{
  __shared__ _Float16 tile[32][33];
  const int t = threadIdx.x;
  const int s0 = blockIdx.x * 32, c0 = blockIdx.y * 32, b = blockIdx.z;
#pragma unroll
  for (int i = 0; i < 4; ++i) {
    int idx = t + 256 * i;
    int cc = idx >> 5, ss = idx & 31;
    tile[cc][ss] = (_Float16)x[((size_t)b * C_ + c0 + cc) * S_ + s0 + ss];
  }
  __syncthreads();
#pragma unroll
  for (int i = 0; i < 4; ++i) {
    int idx = t + 256 * i;
    int ss = idx >> 5, cc = idx & 31;
    xT[((size_t)b * S_ + s0 + ss) * C_ + c0 + cc] = tile[cc][ss];
  }
}

__global__ __launch_bounds__(256) void transpose_w_kernel(
    const float* __restrict__ in, _Float16* __restrict__ outT, int K, int N)
{
  __shared__ _Float16 tile[32][33];
  const int t = threadIdx.x;
  const int n0 = blockIdx.x * 32, k0 = blockIdx.y * 32;
#pragma unroll
  for (int i = 0; i < 4; ++i) {
    int idx = t + 256 * i;
    int kk = idx >> 5, nn = idx & 31;
    tile[kk][nn] = (_Float16)in[(size_t)(k0 + kk) * N + n0 + nn];
  }
  __syncthreads();
#pragma unroll
  for (int i = 0; i < 4; ++i) {
    int idx = t + 256 * i;
    int nn = idx >> 5, kk = idx & 31;
    outT[(size_t)(n0 + nn) * K + k0 + kk] = tile[kk][nn];
  }
}

// ---------------------------------------------------------------------------
// GEMM core: block = 256 threads (8 waves), tile M=128 x N=64, K=256.
// B panel (64x256 f16, pre-transposed [n][k]) staged once into LDS
// (TDM or async path). A fragments load straight from f16 global.
// All 4 B-fragments preloaded so the 4 WMMAs issue without ds waits between.
// ---------------------------------------------------------------------------
template <bool USE_TDM>
__device__ __forceinline__ void gemm_tile_128x64(
    const _Float16* __restrict__ Arows,   // &A[row0*256], row-major f16, ld=256
    const _Float16* __restrict__ BT,      // &B^T[nbase*256], [n][k] f16, ld=256
    _Float16* ldsB, v8f acc[4], int ms, int l15, int hi, int t)
{
  if (USE_TDM) stage_panel_tdm(BT, ldsB, t);
  else         stage_panel_async(BT, ldsB, t);
  __syncthreads();

#pragma unroll
  for (int kb = 0; kb < 256; kb += 32) {
    Frag16 a;
    const _Float16* pa = Arows + (size_t)(ms + l15) * 256 + kb + hi * 16;
    a.h[0] = *(const v8h*)pa;
    a.h[1] = *(const v8h*)(pa + 8);
    Frag16 bf[4];
#pragma unroll
    for (int n = 0; n < 4; ++n) {
      const _Float16* pb = &ldsB[(n * 16 + l15) * LDB + kb + hi * 16];
      bf[n].h[0] = *(const v8h*)pb;
      bf[n].h[1] = *(const v8h*)(pb + 8);
    }
#pragma unroll
    for (int n = 0; n < 4; ++n)
      acc[n] = __builtin_amdgcn_wmma_f32_16x16x32_f16(false, a.v, false, bf[n].v,
                                                      (short)0, acc[n], false, false);
  }
}

// ---------------------------------------------------------------------------
// Kernel 1: qkv = xT @ WqkvT^T + b ; scatter to Qh/Kh [s][hd] and Vt [hd][s].
// grid (S/128, 768/64, B)  -- uses TDM staging
// ---------------------------------------------------------------------------
__global__ __launch_bounds__(256) void qkv_kernel(
    const _Float16* __restrict__ xT, const _Float16* __restrict__ WqkvT,
    const float* __restrict__ bqkv,
    _Float16* __restrict__ Qh, _Float16* __restrict__ Kh, _Float16* __restrict__ Vt)
{
  __shared__ _Float16 ldsB[64 * LDB];
  const int t = threadIdx.x, lane = t & 31, l15 = lane & 15, hi = (lane >> 4) & 1;
  const int w = t >> 5;
  const int sbase = blockIdx.x * 128, nbase = blockIdx.y * 64, b = blockIdx.z;
  const int ms = w * 16;

  v8f acc[4];
  const v8f vz = {};
#pragma unroll
  for (int n = 0; n < 4; ++n) acc[n] = vz;

  gemm_tile_128x64<true>(xT + ((size_t)b * S_ + sbase) * C_,
                         WqkvT + (size_t)nbase * C_, ldsB, acc, ms, l15, hi, t);

#pragma unroll
  for (int n = 0; n < 4; ++n) {
    int nglob = nbase + n * 16 + l15;
    float bias = bqkv[nglob];
    int sec = nglob >> 8, col = nglob & 255;
    int head = col >> 6, hd = col & 63;
    size_t bh = (size_t)(b * NH_ + head);
#pragma unroll
    for (int r = 0; r < 8; ++r) {
      int s = sbase + ms + r + 8 * hi;
      _Float16 val = (_Float16)(acc[n][r] + bias);
      if (sec == 0)      Qh[(bh * S_ + s) * HD_ + hd] = val;
      else if (sec == 1) Kh[(bh * S_ + s) * HD_ + hd] = val;
      else               Vt[(bh * HD_ + hd) * S_ + s] = val;
    }
  }
}

// ---------------------------------------------------------------------------
// Kernel 2: flash attention, one wave per (b, head, 32-row q tile).
// S^T = K.Q^T trick: score C-fragments feed P@V A-fragments directly.
// ---------------------------------------------------------------------------
__global__ __launch_bounds__(128) void attn_kernel(
    const _Float16* __restrict__ Qh, const _Float16* __restrict__ Kh,
    const _Float16* __restrict__ Vt, _Float16* __restrict__ vals)
{
  const int t = threadIdx.x, lane = t & 31, l15 = lane & 15, hi = (lane >> 4) & 1;
  const int w = t >> 5;
  int task = blockIdx.x * 4 + w;          // B*NH*(S/32) = 2048 tasks
  int b    = task >> 9;
  int head = (task >> 7) & 3;
  int qt   = task & 127;
  int sq0  = qt * 32;

  const _Float16* Qp = Qh + (size_t)(b * NH_ + head) * S_ * HD_;
  const _Float16* Kp = Kh + (size_t)(b * NH_ + head) * S_ * HD_;
  const _Float16* Vp = Vt + (size_t)(b * NH_ + head) * HD_ * S_;

  Frag16 qb[2][2];
#pragma unroll
  for (int q = 0; q < 2; ++q)
#pragma unroll
    for (int ch = 0; ch < 2; ++ch) {
      const _Float16* p = Qp + (size_t)(sq0 + q * 16 + l15) * HD_ + ch * 32 + hi * 16;
      qb[q][ch].h[0] = *(const v8h*)p;
      qb[q][ch].h[1] = *(const v8h*)(p + 8);
    }

  const v8f vz = {};
  v8f o[2][4];
#pragma unroll
  for (int q = 0; q < 2; ++q)
#pragma unroll
    for (int n = 0; n < 4; ++n) o[q][n] = vz;
  float m[2] = {-3.0e38f, -3.0e38f}, ssum[2] = {0.f, 0.f};
  const float scale = 0.125f;             // 1/sqrt(64)

  for (int skc = 0; skc < S_; skc += 32) {
    v8f st[2][2];
#pragma unroll
    for (int q = 0; q < 2; ++q)
#pragma unroll
      for (int tt = 0; tt < 2; ++tt) st[q][tt] = vz;
#pragma unroll
    for (int tt = 0; tt < 2; ++tt)
#pragma unroll
      for (int ch = 0; ch < 2; ++ch) {
        Frag16 ak;  // A = K rows
        const _Float16* p = Kp + (size_t)(skc + tt * 16 + l15) * HD_ + ch * 32 + hi * 16;
        ak.h[0] = *(const v8h*)p;
        ak.h[1] = *(const v8h*)(p + 8);
#pragma unroll
        for (int q = 0; q < 2; ++q)
          st[q][tt] = __builtin_amdgcn_wmma_f32_16x16x32_f16(
              false, ak.v, false, qb[q][ch].v, (short)0, st[q][tt], false, false);
      }

    Frag16 pa[2];
#pragma unroll
    for (int q = 0; q < 2; ++q) {
      float mx = -3.0e38f;
#pragma unroll
      for (int r = 0; r < 8; ++r) {
        st[q][0][r] *= scale; st[q][1][r] *= scale;
        mx = fmaxf(mx, fmaxf(st[q][0][r], st[q][1][r]));
      }
      mx = fmaxf(mx, bcast_lane(mx, lane ^ 16));
      float mnew  = fmaxf(m[q], mx);
      float alpha = __expf(m[q] - mnew);
      float psum = 0.f;
#pragma unroll
      for (int r = 0; r < 8; ++r) {
        float p0 = __expf(st[q][0][r] - mnew);
        float p1 = __expf(st[q][1][r] - mnew);
        psum += p0 + p1;
        pa[q].v[r]     = (_Float16)p0;
        pa[q].v[r + 8] = (_Float16)p1;
      }
      psum += bcast_lane(psum, lane ^ 16);
      ssum[q] = ssum[q] * alpha + psum;
      m[q] = mnew;
#pragma unroll
      for (int r = 0; r < 8; ++r) {
        float ar = bcast_lane(alpha, r + 8 * hi);
        o[q][0][r] *= ar; o[q][1][r] *= ar; o[q][2][r] *= ar; o[q][3][r] *= ar;
      }
    }

#pragma unroll
    for (int n = 0; n < 4; ++n) {
      Frag16 vb;
      const _Float16* p = Vp + (size_t)(n * 16 + l15) * S_ + skc + hi * 8;
      vb.h[0] = *(const v8h*)p;
      vb.h[1] = *(const v8h*)(p + 16);
#pragma unroll
      for (int q = 0; q < 2; ++q)
        o[q][n] = __builtin_amdgcn_wmma_f32_16x16x32_f16(
            false, pa[q].v, false, vb.v, (short)0, o[q][n], false, false);
    }
  }

#pragma unroll
  for (int q = 0; q < 2; ++q) {
#pragma unroll
    for (int r = 0; r < 8; ++r) {
      float sr  = bcast_lane(ssum[q], r + 8 * hi);
      float inv = 1.0f / sr;
      int srow  = sq0 + q * 16 + r + 8 * hi;
      size_t base = ((size_t)b * S_ + srow) * C_ + head * HD_;
      vals[base + 0 * 16 + l15] = (_Float16)(o[q][0][r] * inv);
      vals[base + 1 * 16 + l15] = (_Float16)(o[q][1][r] * inv);
      vals[base + 2 * 16 + l15] = (_Float16)(o[q][2][r] * inv);
      vals[base + 3 * 16 + l15] = (_Float16)(o[q][3][r] * inv);
    }
  }
}

// ---------------------------------------------------------------------------
// Kernel 3: out = vals @ Wo + b_o, written transposed to [b][c][s] f32.
// grid (S/128, 256/64, B)  -- uses async-copy staging
// ---------------------------------------------------------------------------
__global__ __launch_bounds__(256) void proj_kernel(
    const _Float16* __restrict__ vals, const _Float16* __restrict__ WoT,
    const float* __restrict__ bo, float* __restrict__ out)
{
  __shared__ _Float16 ldsB[64 * LDB];
  const int t = threadIdx.x, lane = t & 31, l15 = lane & 15, hi = (lane >> 4) & 1;
  const int w = t >> 5;
  const int sbase = blockIdx.x * 128, nbase = blockIdx.y * 64, b = blockIdx.z;
  const int ms = w * 16;

  v8f acc[4];
  const v8f vz = {};
#pragma unroll
  for (int n = 0; n < 4; ++n) acc[n] = vz;

  gemm_tile_128x64<false>(vals + ((size_t)b * S_ + sbase) * C_,
                          WoT + (size_t)nbase * C_, ldsB, acc, ms, l15, hi, t);

#pragma unroll
  for (int n = 0; n < 4; ++n) {
    int cout = nbase + n * 16 + l15;
    float bias = bo[cout];
#pragma unroll
    for (int r = 0; r < 8; ++r) {
      int srow = sbase + ms + r + 8 * hi;
      out[((size_t)b * C_ + cout) * S_ + srow] = acc[n][r] + bias;
    }
  }
}

// ---------------------------------------------------------------------------
extern "C" void kernel_launch(void* const* d_in, const int* in_sizes, int n_in,
                              void* d_out, int out_size, void* d_ws, size_t ws_size,
                              hipStream_t stream) {
  (void)in_sizes; (void)n_in; (void)out_size; (void)ws_size;
  const float* x    = (const float*)d_in[0];
  const float* Wqkv = (const float*)d_in[1];
  const float* bqkv = (const float*)d_in[2];
  const float* Wo   = (const float*)d_in[3];
  const float* bo   = (const float*)d_in[4];
  float* out = (float*)d_out;

  char* ws = (char*)d_ws;
  const size_t qsz = (size_t)B_ * NH_ * S_ * HD_ * sizeof(_Float16);  // 8 MB
  _Float16* Qh    = (_Float16*)(ws);
  _Float16* Kh    = (_Float16*)(ws + qsz);
  _Float16* Vt    = (_Float16*)(ws + 2 * qsz);
  _Float16* vals  = (_Float16*)(ws + 3 * qsz);
  _Float16* xT    = (_Float16*)(ws + 4 * qsz);                        // 8 MB
  _Float16* WqkvT = (_Float16*)(ws + 5 * qsz);                        // 384 KB
  _Float16* WoT   = (_Float16*)(ws + 5 * qsz + (size_t)C3_ * C_ * 2);

  transpose_x_kernel<<<dim3(S_ / 32, C_ / 32, B_), 256, 0, stream>>>(x, xT);
  transpose_w_kernel<<<dim3(C3_ / 32, C_ / 32), 256, 0, stream>>>(Wqkv, WqkvT, C_, C3_);
  transpose_w_kernel<<<dim3(C_ / 32, C_ / 32), 256, 0, stream>>>(Wo, WoT, C_, C_);

  qkv_kernel<<<dim3(S_ / 128, C3_ / 64, B_), 256, 0, stream>>>(xT, WqkvT, bqkv, Qh, Kh, Vt);
  attn_kernel<<<dim3(B_ * NH_ * (S_ / 32) / 4), 128, 0, stream>>>(Qh, Kh, Vt, vals);
  proj_kernel<<<dim3(S_ / 128, C_ / 64, B_), 256, 0, stream>>>(vals, WoT, bo, out);
}
